// Attention_uia_81183471829657
// MI455X (gfx1250) — matrix operand order repaired and checked by reading, not compiled, not run on hardware
//
#include <hip/hip_runtime.h>
#include <hip/hip_bf16.h>

typedef __attribute__((ext_vector_type(16))) __bf16 v16bf;
typedef __attribute__((ext_vector_type(8)))  float  v8f;

#define WMMA_BF16(a, b, c) \
  __builtin_amdgcn_wmma_f32_16x16x32_bf16(false, (a), false, (b), (short)0, (c), false, false)

static constexpr int BATCH = 4;
static constexpr int SEQ   = 2048;
static constexpr int DIM   = 1024;
static constexpr int HEADS = 16;
static constexpr int HD    = 64;
static constexpr float SCALE = 0.125f;           // 64^-0.5
static constexpr int TOKENS = BATCH * SEQ;       // 8192

// ---------------- fragment loaders ----------------
// A-frag (16x32 bf16): lane l<16: M=l, K 0..15; l>=16: M=l-16,
// K 16..31 contiguous. Caller passes pointer at (row, kbase + 16*half).
__device__ __forceinline__ v16bf a_frag_from_f32(const float* __restrict__ p) {
  v16bf r;
#pragma unroll
  for (int j = 0; j < 8; ++j) { r[j] = (__bf16)p[j]; r[j + 8] = (__bf16)p[j + 8]; }
  return r;
}
__device__ __forceinline__ v16bf a_frag_from_bf16(const __bf16* __restrict__ p) {
  v16bf r;
#pragma unroll
  for (int j = 0; j < 8; ++j) { r[j] = p[j]; r[j + 8] = p[j + 8]; }
  return r;
}
// B-frag (32x16 bf16): lane l holds column n=l&15, K = 16*(l>>4) + 0..15 contiguous.
__device__ __forceinline__ v16bf b_frag_from_f32(const float* __restrict__ p) {
  v16bf r;
#pragma unroll
  for (int j = 0; j < 16; ++j) r[j] = (__bf16)p[j];
  return r;
}
__device__ __forceinline__ v16bf b_frag_from_bf16(const __bf16* __restrict__ p) {
  v16bf r;
#pragma unroll
  for (int j = 0; j < 16; ++j) r[j] = p[j];
  return r;
}

// ---------------- kernel 1: QKV GEMM (x @ w_qkv^T), 64x64 tile/wave ----------------
// Writes Q*SCALE and K as [B,H,N,hd] bf16, V transposed [B,H,hd,N] bf16.
__global__ __launch_bounds__(32)
void qkv_gemm_kernel(const float* __restrict__ x, const float* __restrict__ w,
                     __bf16* __restrict__ qo, __bf16* __restrict__ ko,
                     __bf16* __restrict__ vto) {
  const int lane = threadIdx.x;
  const int half = lane >> 4, lm = lane & 15;
  const int m0 = blockIdx.x * 64;
  const int n0 = blockIdx.y * 64;

  v8f acc[4][4];
#pragma unroll
  for (int i = 0; i < 4; ++i)
#pragma unroll
    for (int t = 0; t < 4; ++t) { v8f z = {}; acc[i][t] = z; }

  for (int kk = 0; kk < DIM; kk += 32) {
    v16bf af[4];
#pragma unroll
    for (int i = 0; i < 4; ++i)
      af[i] = a_frag_from_f32(x + (size_t)(m0 + 16 * i + lm) * DIM + kk + 16 * half);
#pragma unroll
    for (int t = 0; t < 4; ++t) {
      v16bf b = b_frag_from_f32(w + (size_t)(n0 + 16 * t + lm) * DIM + kk + 16 * half);
#pragma unroll
      for (int i = 0; i < 4; ++i)
        acc[i][t] = WMMA_BF16(af[i], b, acc[i][t]);
    }
  }

  // C layout: VGPR r, lane l -> M = r + 8*(l>>4), N = l&15
#pragma unroll
  for (int i = 0; i < 4; ++i) {
#pragma unroll
    for (int t = 0; t < 4; ++t) {
      const int d = n0 + 16 * t + lm;
#pragma unroll
      for (int r = 0; r < 8; ++r) {
        const int row = m0 + 16 * i + r + 8 * half;   // global token
        const int b = row >> 11, n = row & (SEQ - 1);
        const float val = acc[i][t][r];
        if (d < DIM) {                                 // Q (pre-scaled)
          const int h = d >> 6, e = d & 63;
          qo[(((size_t)(b * HEADS + h) * SEQ) + n) * HD + e] = (__bf16)(val * SCALE);
        } else if (d < 2 * DIM) {                      // K
          const int c = d - DIM, h = c >> 6, e = c & 63;
          ko[(((size_t)(b * HEADS + h) * SEQ) + n) * HD + e] = (__bf16)val;
        } else {                                       // V transposed [B,H,hd,N]
          const int c = d - 2 * DIM, h = c >> 6, e = c & 63;
          vto[(((size_t)(b * HEADS + h) * HD) + e) * SEQ + n] = (__bf16)val;
        }
      }
    }
  }
}

// ---------------- kernel 2: flash attention, 32 query rows/wave ----------------
__global__ __launch_bounds__(32)
void flash_attn_kernel(const __bf16* __restrict__ q, const __bf16* __restrict__ k,
                       const __bf16* __restrict__ vt, __bf16* __restrict__ o) {
  __shared__ __bf16 ptile[32 * 32];   // 2 row-tiles x 32 keys

  const int lane = threadIdx.x;
  const int half = lane >> 4, lm = lane & 15;
  const int bh = blockIdx.x;               // 0..B*H-1
  const int q0 = blockIdx.y * 32;          // first of 32 query rows

  const __bf16* qp = q  + (size_t)bh * SEQ * HD;
  const __bf16* kp = k  + (size_t)bh * SEQ * HD;
  const __bf16* vp = vt + (size_t)bh * HD * SEQ;

  // Q A-frags: row tile u (rows q0+16u..), K-dim step s (0..31 / 32..63)
  v16bf qa[2][2];
#pragma unroll
  for (int u = 0; u < 2; ++u)
#pragma unroll
    for (int s = 0; s < 2; ++s)
      qa[u][s] = a_frag_from_bf16(qp + (size_t)(q0 + 16 * u + lm) * HD + 32 * s + 16 * half);

  v8f ov[2][4];
#pragma unroll
  for (int u = 0; u < 2; ++u)
#pragma unroll
    for (int t = 0; t < 4; ++t) { v8f z = {}; ov[u][t] = z; }
  float m_i[2][8], l_i[2][8];
#pragma unroll
  for (int u = 0; u < 2; ++u)
#pragma unroll
    for (int r = 0; r < 8; ++r) { m_i[u][r] = -3.0e38f; l_i[u][r] = 0.0f; }

  for (int j = 0; j < SEQ; j += 32) {
    if (j + 32 < SEQ) {   // prefetch next K block (lowers to global_prefetch_b8)
      __builtin_prefetch(kp + (size_t)(j + 32 + lane) * HD, 0, 1);
    }

    // scores: 2 row-tiles x 2 key-tiles, hd=64 -> 2 chained WMMAs each
    v8f s[2][2];
#pragma unroll
    for (int c = 0; c < 2; ++c) {
      v16bf kb0 = b_frag_from_bf16(kp + (size_t)(j + 16 * c + lm) * HD + 16 * half);
      v16bf kb1 = b_frag_from_bf16(kp + (size_t)(j + 16 * c + lm) * HD + 32 + 16 * half);
#pragma unroll
      for (int u = 0; u < 2; ++u) {
        v8f acc = {};
        acc = WMMA_BF16(qa[u][0], kb0, acc);
        acc = WMMA_BF16(qa[u][1], kb1, acc);
        s[u][c] = acc;
      }
    }

    // online softmax per row M = 16u + r + 8*half (row spread across 16 lanes)
#pragma unroll
    for (int u = 0; u < 2; ++u) {
#pragma unroll
      for (int r = 0; r < 8; ++r) {
        float v0 = s[u][0][r], v1 = s[u][1][r];
        float mx = fmaxf(v0, v1);
#pragma unroll
        for (int off = 8; off >= 1; off >>= 1) mx = fmaxf(mx, __shfl_xor(mx, off, 16));
        const float mnew  = fmaxf(m_i[u][r], mx);
        const float alpha = __expf(m_i[u][r] - mnew);
        const float p0 = __expf(v0 - mnew);
        const float p1 = __expf(v1 - mnew);
        float rs = p0 + p1;
#pragma unroll
        for (int off = 8; off >= 1; off >>= 1) rs += __shfl_xor(rs, off, 16);
        l_i[u][r] = l_i[u][r] * alpha + rs;
        m_i[u][r] = mnew;
#pragma unroll
        for (int t = 0; t < 4; ++t) ov[u][t][r] *= alpha;
        const int row = 16 * u + r + 8 * half;
        ptile[row * 32 + lm]      = (__bf16)p0;
        ptile[row * 32 + 16 + lm] = (__bf16)p1;
      }
    }

    // P tiles from LDS into A-frag layout
    v16bf pa[2];
#pragma unroll
    for (int u = 0; u < 2; ++u)
      pa[u] = a_frag_from_bf16(&ptile[(16 * u + lm) * 32 + 16 * half]);

    // O += P @ V  (V^T layout -> contiguous key runs); V frags shared by both row tiles
#pragma unroll
    for (int t = 0; t < 4; ++t) {
      v16bf vb = b_frag_from_bf16(vp + (size_t)(16 * t + lm) * SEQ + j + 16 * half);
#pragma unroll
      for (int u = 0; u < 2; ++u)
        ov[u][t] = WMMA_BF16(pa[u], vb, ov[u][t]);
    }
  }

  // epilogue: normalize, write [B,N,C] bf16
  const int b = bh >> 4, h = bh & (HEADS - 1);
#pragma unroll
  for (int u = 0; u < 2; ++u) {
#pragma unroll
    for (int t = 0; t < 4; ++t) {
#pragma unroll
      for (int r = 0; r < 8; ++r) {
        const int row = q0 + 16 * u + r + 8 * half;
        const float val = ov[u][t][r] / l_i[u][r];
        o[((size_t)(b * SEQ + row)) * DIM + h * HD + 16 * t + lm] = (__bf16)val;
      }
    }
  }
}

// ---------------- kernel 3: proj GEMM (attn @ w_proj^T + bias), 64x64 tile ----------------
__global__ __launch_bounds__(32)
void proj_gemm_kernel(const __bf16* __restrict__ a, const float* __restrict__ w,
                      const float* __restrict__ bias, float* __restrict__ out) {
  const int lane = threadIdx.x;
  const int half = lane >> 4, lm = lane & 15;
  const int m0 = blockIdx.x * 64;
  const int n0 = blockIdx.y * 64;

  v8f acc[4][4];
#pragma unroll
  for (int i = 0; i < 4; ++i)
#pragma unroll
    for (int t = 0; t < 4; ++t) { v8f z = {}; acc[i][t] = z; }

  for (int kk = 0; kk < DIM; kk += 32) {
    v16bf af[4];
#pragma unroll
    for (int i = 0; i < 4; ++i)
      af[i] = a_frag_from_bf16(a + (size_t)(m0 + 16 * i + lm) * DIM + kk + 16 * half);
#pragma unroll
    for (int t = 0; t < 4; ++t) {
      v16bf b = b_frag_from_f32(w + (size_t)(n0 + 16 * t + lm) * DIM + kk + 16 * half);
#pragma unroll
      for (int i = 0; i < 4; ++i)
        acc[i][t] = WMMA_BF16(af[i], b, acc[i][t]);
    }
  }

#pragma unroll
  for (int i = 0; i < 4; ++i) {
#pragma unroll
    for (int t = 0; t < 4; ++t) {
      const int d = n0 + 16 * t + lm;
      const float bv = bias[d];
#pragma unroll
      for (int r = 0; r < 8; ++r) {
        const int row = m0 + 16 * i + r + 8 * half;
        out[(size_t)row * DIM + d] = acc[i][t][r] + bv;
      }
    }
  }
}

// ---------------- launcher ----------------
extern "C" void kernel_launch(void* const* d_in, const int* in_sizes, int n_in,
                              void* d_out, int out_size, void* d_ws, size_t ws_size,
                              hipStream_t stream) {
  const float* x      = (const float*)d_in[0];   // [4,2048,1024]
  const float* w_qkv  = (const float*)d_in[1];   // [3072,1024]
  const float* w_proj = (const float*)d_in[2];   // [1024,1024]
  const float* b_proj = (const float*)d_in[3];   // [1024]
  float* out = (float*)d_out;                    // [4,2048,1024]

  const size_t PER = (size_t)BATCH * HEADS * SEQ * HD;  // 8,388,608 elems
  __bf16* qb  = (__bf16*)d_ws;
  __bf16* kb  = qb  + PER;
  __bf16* vtb = kb  + PER;
  __bf16* ab  = vtb + PER;   // attention output [B,N,C] bf16

  // 1) QKV: M=8192 (x64), Nout=3072 (x64)
  qkv_gemm_kernel<<<dim3(TOKENS / 64, (3 * DIM) / 64), 32, 0, stream>>>(
      x, w_qkv, qb, kb, vtb);

  // 2) flash attention: (B*H) x (SEQ/32) waves
  flash_attn_kernel<<<dim3(BATCH * HEADS, SEQ / 32), 32, 0, stream>>>(
      qb, kb, vtb, ab);

  // 3) projection: M=8192 (x64), Nout=1024 (x64)
  proj_gemm_kernel<<<dim3(TOKENS / 64, DIM / 64), 32, 0, stream>>>(
      ab, w_proj, b_proj, out);
}
